// DistBiasSelfAttention_64493228917282
// MI455X (gfx1250) — matrix operand order repaired and checked by reading, not compiled, not run on hardware
//
#include <hip/hip_runtime.h>

// Problem constants (from reference): B=4, Q=1024, C=256, H=8, D=32
#define BATCH 4
#define QLEN  1024
#define CDIM  256
#define HEADS 8
#define HD    32
#define C3    (3 * CDIM)
#define BQ    (BATCH * QLEN)

typedef __attribute__((ext_vector_type(16))) __bf16 v16bf;
typedef __attribute__((ext_vector_type(8)))  float  v8f;

// ---------------------------------------------------------------------------
// WMMA wrapper: D = A(16x32 bf16) * B(32x16 bf16) + C(16x16 f32)
// ---------------------------------------------------------------------------
__device__ __forceinline__ v8f wmma_bf16(v16bf a, v16bf b, v8f c) {
  return __builtin_amdgcn_wmma_f32_16x16x32_bf16(
      /*neg_a=*/false, a, /*neg_b=*/false, b,
      /*c_mod=*/(short)0, c, /*reuse_a=*/false, /*reuse_b=*/false);
}

// ---------------------------------------------------------------------------
// Fragment loaders (wave32; contiguous k order, not the ISA table order)
// A 16x32 bf16: lane m=l&15 holds row m; lanes 0-15 K=0..15,
//               lanes 16-31 K=16..31; 2 K-values packed per VGPR.
// B 32x16 bf16: lane n=l&15 holds col n; lanes 0-15 K=0..15, 16-31 K=16..31.
// C/D 16x16 f32: VGPR v holds row v+8*(l>>4), col l&15.
// ---------------------------------------------------------------------------
__device__ __forceinline__ v16bf load_a_f32(const float* __restrict__ src,
                                            int ld, int lane) {
  const int m = lane & 15, g = lane >> 4;
  const float* row = src + (size_t)m * ld;
  v16bf a;
#pragma unroll
  for (int v = 0; v < 8; ++v) {
    const int k = 16 * g + 2 * v;
    a[2 * v]     = (__bf16)row[k];
    a[2 * v + 1] = (__bf16)row[k + 1];
  }
  return a;
}

__device__ __forceinline__ v16bf load_a_lds(const __bf16* src, int ld, int lane) {
  const int m = lane & 15, g = lane >> 4;
  const __bf16* row = src + m * ld;
  v16bf a;
#pragma unroll
  for (int v = 0; v < 8; ++v) {
    const int k = 16 * g + 2 * v;
    a[2 * v]     = row[k];
    a[2 * v + 1] = row[k + 1];
  }
  return a;
}

// fragment element (k, n) = src[n*ld + k]   (B = W^T with W row-major [N,K])
__device__ __forceinline__ v16bf load_b_colmajor(const float* src, int ld,
                                                 int lane) {
  const int n = lane & 15, g = lane >> 4;
  const float* p = src + (size_t)n * ld + 16 * g;
  v16bf b;
#pragma unroll
  for (int v = 0; v < 8; ++v) {
    b[2 * v]     = (__bf16)p[2 * v];
    b[2 * v + 1] = (__bf16)p[2 * v + 1];
  }
  return b;
}

// fragment element (k, n) = src[k*ld + n]   (V tile: rows = k, cols = d)
__device__ __forceinline__ v16bf load_b_rowmajor(const float* src, int ld,
                                                 int lane) {
  const int n = lane & 15, g = lane >> 4;
  const float* p = src + (size_t)(16 * g) * ld + n;
  v16bf b;
#pragma unroll
  for (int v = 0; v < 8; ++v) {
    b[2 * v]     = (__bf16)p[(size_t)(2 * v) * ld];
    b[2 * v + 1] = (__bf16)p[(size_t)(2 * v + 1) * ld];
  }
  return b;
}

// ---------------------------------------------------------------------------
// Kernel 1: fused QKV projection.  qkv[BQ, 3C] = feats[BQ, C] @ W^T + b
// ---------------------------------------------------------------------------
__global__ __launch_bounds__(32) void qkv_gemm_kernel(
    const float* __restrict__ A, const float* __restrict__ W,
    const float* __restrict__ bias, float* __restrict__ out) {
  const int lane = threadIdx.x;
  const int nt = blockIdx.x;  // 0..47  (N = 768)
  const int mt = blockIdx.y;  // 0..255 (M = 4096)
  const float* aptr = A + (size_t)(mt * 16) * CDIM;
  const float* wptr = W + (size_t)(nt * 16) * CDIM;
  v8f acc = {};
#pragma unroll
  for (int kk = 0; kk < CDIM; kk += 32) {
    v16bf af = load_a_f32(aptr + kk, CDIM, lane);
    v16bf bf = load_b_colmajor(wptr + kk, CDIM, lane);
    acc = wmma_bf16(af, bf, acc);
  }
  const int n = lane & 15, g = lane >> 4;
  const float bn = bias[nt * 16 + n];
#pragma unroll
  for (int v = 0; v < 8; ++v) {
    const int r = mt * 16 + v + 8 * g;
    out[(size_t)r * C3 + nt * 16 + n] = acc[v] + bn;
  }
}

// ---------------------------------------------------------------------------
// Kernel 2: tauS[bq, h] = (feat[bq,:] . tau_w[h,:] + tau_b[h]) * scale[h]
// ---------------------------------------------------------------------------
__global__ __launch_bounds__(256) void tau_kernel(
    const float* __restrict__ feats, const float* __restrict__ tau_w,
    const float* __restrict__ tau_b, const float* __restrict__ scale,
    float* __restrict__ tauS) {
  const int i = blockIdx.x * 256 + threadIdx.x;  // 0..BQ*H-1
  const int bq = i >> 3, h = i & 7;
  const float* f = feats + (size_t)bq * CDIM;
  const float* w = tau_w + (size_t)h * CDIM;
  float s = 0.f;
#pragma unroll 4
  for (int k = 0; k < CDIM; k += 4) {
    s += f[k] * w[k] + f[k + 1] * w[k + 1] + f[k + 2] * w[k + 2] +
         f[k + 3] * w[k + 3];
  }
  tauS[i] = (s + tau_b[h]) * scale[h];
}

// ---------------------------------------------------------------------------
// Kernel 3: flash attention with distance bias.
// Block = 4 waves, all on the same (b, h); wave w owns q-tile (4*bx + w).
// Per 32-column k step, the K and V tiles (32x32 f32 each, 4 KB) are staged
// once per block into LDS with async global->LDS DMA (ASYNCcnt), then all
// four waves build WMMA fragments from LDS.
// ---------------------------------------------------------------------------
__global__ __launch_bounds__(128) void attn_kernel(
    const float* __restrict__ qkv, const float* __restrict__ xyz,
    const float* __restrict__ tauS, float* __restrict__ ctx) {
  __shared__ float kv_k[32 * 32];       // K tile: [kcol][d]
  __shared__ float kv_v[32 * 32];       // V tile: [krow][d]
  __shared__ __bf16 lds_p[4][16 * 32];  // per-wave P-tile staging

  const int wave = threadIdx.x >> 5;
  const int lane = threadIdx.x & 31;
  const int qt = blockIdx.x * 4 + wave;  // 0..63
  const int h  = blockIdx.y;
  const int b  = blockIdx.z;
  const int n = lane & 15;
  const int g = lane >> 4;
  const int qbase = b * QLEN + qt * 16;

  // Q fragment for this wave's tile (rows qbase..qbase+15, head-h q slice)
  const v16bf Aq = load_a_f32(qkv + (size_t)qbase * C3 + h * HD, C3, lane);

  // Per-lane row state: this lane owns rows r = v + 8*g of C/D fragments.
  float tau_r[8], xq0[8], xq1[8], xq2[8];
#pragma unroll
  for (int v = 0; v < 8; ++v) {
    const int r = qbase + v + 8 * g;
    tau_r[v] = tauS[r * HEADS + h];
    xq0[v] = xyz[r * 3 + 0];
    xq1[v] = xyz[r * 3 + 1];
    xq2[v] = xyz[r * 3 + 2];
  }

  float m_i[8], l_i[8];
  v8f O0 = {};
  v8f O1 = {};
#pragma unroll
  for (int v = 0; v < 8; ++v) { m_i[v] = -1e30f; l_i[v] = 0.0f; }

  const float inv_sqrt_d = 0.17677669529663689f;  // 32^-0.5

  for (int kk = 0; kk < QLEN; kk += 32) {
    // ---- stage K/V tiles into LDS via async DMA (16B per lane per issue)
    {
      const float* kg = qkv + (size_t)(b * QLEN + kk) * C3 + CDIM + h * HD;
      const float* vg = kg + CDIM;
      for (int c = threadIdx.x; c < 256; c += 128) {
        const int row = c >> 3;            // tile row 0..31
        const int off = (c & 7) << 2;      // float offset within row, 16B chunk
        const float* gk = kg + (size_t)row * C3 + off;
        const float* gv = vg + (size_t)row * C3 + off;
        const unsigned lk = (unsigned)(size_t)(const void*)&kv_k[c << 2];
        const unsigned lv = (unsigned)(size_t)(const void*)&kv_v[c << 2];
        asm volatile("global_load_async_to_lds_b128 %0, %1, off"
                     :: "v"(lk), "v"(gk) : "memory");
        asm volatile("global_load_async_to_lds_b128 %0, %1, off"
                     :: "v"(lv), "v"(gv) : "memory");
      }
      asm volatile("s_wait_asynccnt 0x0" ::: "memory");
      __syncthreads();
    }

    // ---- scores: two 16x16 WMMA tiles over k columns [kk, kk+32)
    // K fragment element (d, n) = kv_k[n*32 + d]  (col-major, ld = 32)
    const v16bf K0 = load_b_colmajor(kv_k, 32, lane);
    const v16bf K1 = load_b_colmajor(kv_k + 16 * 32, 32, lane);
    const v8f zero = {};
    v8f S0 = wmma_bf16(Aq, K0, zero);
    v8f S1 = wmma_bf16(Aq, K1, zero);

    // xyz for the two k columns this lane owns
    const int c0 = b * QLEN + kk + n;
    const float xk0x = xyz[c0 * 3 + 0], xk0y = xyz[c0 * 3 + 1],
                xk0z = xyz[c0 * 3 + 2];
    const int c1 = c0 + 16;
    const float xk1x = xyz[c1 * 3 + 0], xk1y = xyz[c1 * 3 + 1],
                xk1z = xyz[c1 * 3 + 2];

    // ---- fp32 bias + online softmax (row reductions across 16-lane halves)
#pragma unroll
    for (int v = 0; v < 8; ++v) {
      float dx = xq0[v] - xk0x, dy = xq1[v] - xk0y, dz = xq2[v] - xk0z;
      const float d0 = __builtin_sqrtf(dx * dx + dy * dy + dz * dz);
      const float s0 = S0[v] * inv_sqrt_d - d0 * tau_r[v];
      dx = xq0[v] - xk1x; dy = xq1[v] - xk1y; dz = xq2[v] - xk1z;
      const float d1 = __builtin_sqrtf(dx * dx + dy * dy + dz * dz);
      const float s1 = S1[v] * inv_sqrt_d - d1 * tau_r[v];

      float t = fmaxf(s0, s1);
      t = fmaxf(t, __shfl_xor(t, 1, 32));
      t = fmaxf(t, __shfl_xor(t, 2, 32));
      t = fmaxf(t, __shfl_xor(t, 4, 32));
      t = fmaxf(t, __shfl_xor(t, 8, 32));
      const float mn = fmaxf(m_i[v], t);
      const float alpha = __expf(m_i[v] - mn);
      const float e0 = __expf(s0 - mn);
      const float e1 = __expf(s1 - mn);
      float rs = e0 + e1;
      rs += __shfl_xor(rs, 1, 32);
      rs += __shfl_xor(rs, 2, 32);
      rs += __shfl_xor(rs, 4, 32);
      rs += __shfl_xor(rs, 8, 32);
      l_i[v] = l_i[v] * alpha + rs;
      m_i[v] = mn;
      O0[v] *= alpha;
      O1[v] *= alpha;

      // stage P (bf16) in LDS for D->A layout conversion (per-wave region)
      const int r = v + 8 * g;
      lds_p[wave][r * 32 + n]      = (__bf16)e0;
      lds_p[wave][r * 32 + 16 + n] = (__bf16)e1;
    }
    // LDS ops are in-order within a wave; stop compiler reordering only.
    asm volatile("" ::: "memory");

    const v16bf P = load_a_lds(&lds_p[wave][0], 32, lane);

    // ---- O += P @ V   (V fragment element (k', n) = kv_v[k'*32 + dcol])
    const v16bf V0 = load_b_rowmajor(kv_v, 32, lane);
    const v16bf V1 = load_b_rowmajor(kv_v + 16, 32, lane);
    O0 = wmma_bf16(P, V0, O0);
    O1 = wmma_bf16(P, V1, O1);

    __syncthreads();  // protect kv_k/kv_v before next iteration's staging
  }

  // ---- normalize and write context [BQ, C]
#pragma unroll
  for (int v = 0; v < 8; ++v) {
    const float inv = 1.0f / l_i[v];
    const int r = qbase + v + 8 * g;
    ctx[(size_t)r * CDIM + h * HD + n]      = O0[v] * inv;
    ctx[(size_t)r * CDIM + h * HD + 16 + n] = O1[v] * inv;
  }
}

// ---------------------------------------------------------------------------
// Kernel 4: out projection + bias + residual.  x = ctx @ out_w^T + out_b + feats
// ---------------------------------------------------------------------------
__global__ __launch_bounds__(32) void out_gemm_kernel(
    const float* __restrict__ ctx, const float* __restrict__ W,
    const float* __restrict__ bias, const float* __restrict__ feats,
    float* __restrict__ xout) {
  const int lane = threadIdx.x;
  const int nt = blockIdx.x;  // 0..15
  const int mt = blockIdx.y;  // 0..255
  const float* aptr = ctx + (size_t)(mt * 16) * CDIM;
  const float* wptr = W + (size_t)(nt * 16) * CDIM;
  v8f acc = {};
#pragma unroll
  for (int kk = 0; kk < CDIM; kk += 32) {
    v16bf af = load_a_f32(aptr + kk, CDIM, lane);
    v16bf bf = load_b_colmajor(wptr + kk, CDIM, lane);
    acc = wmma_bf16(af, bf, acc);
  }
  const int n = lane & 15, g = lane >> 4;
  const int c = nt * 16 + n;
  const float bn = bias[c];
#pragma unroll
  for (int v = 0; v < 8; ++v) {
    const int r = mt * 16 + v + 8 * g;
    xout[(size_t)r * CDIM + c] = acc[v] + bn + feats[(size_t)r * CDIM + c];
  }
}

// ---------------------------------------------------------------------------
// Kernel 5: LayerNorm over C=256.  One wave per row.
// ---------------------------------------------------------------------------
__global__ __launch_bounds__(256) void ln_kernel(
    const float* __restrict__ x, const float* __restrict__ gamma,
    const float* __restrict__ beta, float* __restrict__ out) {
  const int wave = threadIdx.x >> 5;
  const int lane = threadIdx.x & 31;
  const int row = blockIdx.x * 8 + wave;
  const float* xr = x + (size_t)row * CDIM;
  float vals[8];
  float s = 0.f, s2 = 0.f;
#pragma unroll
  for (int j = 0; j < 8; ++j) {
    const float t = xr[lane + 32 * j];
    vals[j] = t;
    s += t;
    s2 += t * t;
  }
#pragma unroll
  for (int mask = 1; mask < 32; mask <<= 1) {
    s  += __shfl_xor(s, mask, 32);
    s2 += __shfl_xor(s2, mask, 32);
  }
  const float mu = s * (1.0f / CDIM);
  const float var = fmaxf(s2 * (1.0f / CDIM) - mu * mu, 0.0f);
  const float inv = rsqrtf(var + 1e-5f);
  float* orow = out + (size_t)row * CDIM;
#pragma unroll
  for (int j = 0; j < 8; ++j) {
    const int c = lane + 32 * j;
    orow[c] = (vals[j] - mu) * inv * gamma[c] + beta[c];
  }
}

// ---------------------------------------------------------------------------
// Launch
// ---------------------------------------------------------------------------
extern "C" void kernel_launch(void* const* d_in, const int* in_sizes, int n_in,
                              void* d_out, int out_size, void* d_ws,
                              size_t ws_size, hipStream_t stream) {
  (void)in_sizes; (void)n_in; (void)out_size; (void)ws_size;
  const float* feats     = (const float*)d_in[0];
  const float* xyz       = (const float*)d_in[1];
  const float* in_proj_w = (const float*)d_in[2];
  const float* in_proj_b = (const float*)d_in[3];
  const float* out_w     = (const float*)d_in[4];
  const float* out_b     = (const float*)d_in[5];
  const float* tau_w     = (const float*)d_in[6];
  const float* tau_b     = (const float*)d_in[7];
  const float* scale     = (const float*)d_in[8];
  const float* gamma     = (const float*)d_in[9];
  const float* beta      = (const float*)d_in[10];
  float* outp = (float*)d_out;

  // Workspace carve-up (floats): qkv | tauS | ctx | xbuf  (~21.1 MB total)
  float* ws   = (float*)d_ws;
  float* qkv  = ws;                                   // BQ * 3C
  float* tauS = qkv + (size_t)BQ * C3;                // BQ * H
  float* ctx  = tauS + (size_t)BQ * HEADS;            // BQ * C
  float* xbuf = ctx + (size_t)BQ * CDIM;              // BQ * C

  qkv_gemm_kernel<<<dim3(C3 / 16, BQ / 16), 32, 0, stream>>>(
      feats, in_proj_w, in_proj_b, qkv);
  tau_kernel<<<dim3((BQ * HEADS) / 256), 256, 0, stream>>>(
      feats, tau_w, tau_b, scale, tauS);
  attn_kernel<<<dim3(QLEN / 16 / 4, HEADS, BATCH), 128, 0, stream>>>(
      qkv, xyz, tauS, ctx);
  out_gemm_kernel<<<dim3(CDIM / 16, BQ / 16), 32, 0, stream>>>(
      ctx, out_w, out_b, feats, xbuf);
  ln_kernel<<<dim3(BQ / 8), 256, 0, stream>>>(xbuf, gamma, beta, outp);
}
